// ODELSTMCell_1486058684809
// MI455X (gfx1250) — compile-verified, matrix operand order checked by reading, not run on hardware
//
#include <hip/hip_runtime.h>
#include <math.h>

typedef __bf16 bf16_t;
typedef __attribute__((ext_vector_type(16))) __bf16 v16bf;
typedef __attribute__((ext_vector_type(8)))  __bf16 v8bf;
typedef __attribute__((ext_vector_type(8)))  float  v8f;
typedef __attribute__((ext_vector_type(2)))  float  v2f;

#define BSZ   256
#define INDIM 128
#define HDIM  512
#define STRIP 16
#define BLK   512   // 16 wave32s

// ---------- helpers ----------
__device__ __forceinline__ __bf16 f2bf(float f) {
  unsigned u = __builtin_bit_cast(unsigned, f);
  unsigned r = u + 0x7FFFu + ((u >> 16) & 1u);           // RNE
  unsigned short h = (unsigned short)(r >> 16);
  return __builtin_bit_cast(__bf16, h);
}
__device__ __forceinline__ unsigned pack2bf(float a, float b) {
  unsigned ua = __builtin_bit_cast(unsigned, a);
  unsigned ub = __builtin_bit_cast(unsigned, b);
  ua = ua + 0x7FFFu + ((ua >> 16) & 1u);
  ub = ub + 0x7FFFu + ((ub >> 16) & 1u);
  return (ua >> 16) | (ub & 0xFFFF0000u);
}
__device__ __forceinline__ float sigm(float x) { return 1.0f / (1.0f + __expf(-x)); }
__device__ __forceinline__ v8f zero8() { v8f z = {0.f,0.f,0.f,0.f,0.f,0.f,0.f,0.f}; return z; }

// A fragment (16x32 bf16, ISA 7.12.2) from an LDS bf16 strip: lane holds row m=L%16,
// K = {kb..kb+7, kb+16..kb+23} -> two ds_load_b128, zero VALU.
__device__ __forceinline__ v16bf frag_a_bf(const bf16_t* A, int lda, int m, int kb) {
  const v8bf* p = (const v8bf*)(A + (size_t)m * lda + kb);   // 16B aligned
  v8bf lo = p[0];
  v8bf hi = p[2];
  v16bf r;
#pragma unroll
  for (int i = 0; i < 8; ++i) { r[i] = lo[i]; r[8 + i] = hi[i]; }
  return r;
}
// B fragment (32x16 bf16): lane holds output column n=L%16 (a row of W, y = x@W^T)
__device__ __forceinline__ v16bf frag_b(const bf16_t* W, int ldk, int col, int kb) {
  const bf16_t* p = W + (size_t)col * ldk + kb;
  v8bf lo = *(const v8bf*)(p);
  v8bf hi = *(const v8bf*)(p + 16);
  v16bf r;
#pragma unroll
  for (int i = 0; i < 8; ++i) { r[i] = lo[i]; r[8 + i] = hi[i]; }
  return r;
}

// Two output tiles per wave share one A fragment: 1x A load feeds 2 WMMAs.
__device__ __forceinline__ void gemm_pair(const bf16_t* Abf, int lda, int m,
                                          const bf16_t* W, int ldw,
                                          int col0, int col1, int kb_off, int nks,
                                          v8f& acc0, v8f& acc1) {
#pragma unroll 4
  for (int ks = 0; ks < nks; ++ks) {
    int kb = ks * 32 + kb_off;
    v16bf a  = frag_a_bf(Abf, lda, m, kb);
    v16bf b0 = frag_b(W, ldw, col0, kb);
    v16bf b1 = frag_b(W, ldw, col1, kb);
    acc0 = __builtin_amdgcn_wmma_f32_16x16x32_bf16(false, a, false, b0, (short)0, acc0,
                                                   false, false);
    acc1 = __builtin_amdgcn_wmma_f32_16x16x32_bf16(false, a, false, b1, (short)0, acc1,
                                                   false, false);
  }
}

// ---------- kernel 1: bitonic sort of ts -> dts[255], snapshot schedule snap[256] ----------
__global__ void __launch_bounds__(256) sort_ts_kernel(const float* __restrict__ ts,
                                                      float* __restrict__ dts,
                                                      int* __restrict__ snap) {
  __shared__ float v[BSZ];
  __shared__ int   ix[BSZ];
  int t = threadIdx.x;
  v[t] = ts[t]; ix[t] = t;
  __syncthreads();
  for (int k = 2; k <= BSZ; k <<= 1) {
    for (int j = k >> 1; j > 0; j >>= 1) {
      int p = t ^ j;
      if (p > t) {
        bool up = ((t & k) == 0);
        float a = v[t], b = v[p];
        if (up ? (a > b) : (a < b)) {
          v[t] = b; v[p] = a;
          int q = ix[t]; ix[t] = ix[p]; ix[p] = q;
        }
      }
      __syncthreads();
    }
  }
  snap[t] = ix[t];                       // row i snapshotted at state index indices[i]
  if (t < BSZ - 1) {
    float s0 = v[t]     + 1e-4f * ((float)t       / 255.0f);
    float s1 = v[t + 1] + 1e-4f * ((float)(t + 1) / 255.0f);
    dts[t] = s1 - s0;
  }
}

// ---------- kernel 2: convert weights to bf16 (once; L2-resident afterwards) ----------
__global__ void __launch_bounds__(256) convert_kernel(
    const float* __restrict__ Wih, const float* __restrict__ Whh,
    const float* __restrict__ W1,  const float* __restrict__ W2,
    bf16_t* __restrict__ o_ih, bf16_t* __restrict__ o_hh,
    bf16_t* __restrict__ o1,   bf16_t* __restrict__ o2) {
  const int n_ih = 4 * HDIM * INDIM;     // 262144
  const int n_hh = 4 * HDIM * HDIM;      // 1048576
  const int n_w  = HDIM * HDIM;          // 262144
  for (int t = blockIdx.x * blockDim.x + threadIdx.x; t < n_hh;
       t += gridDim.x * blockDim.x) {
    o_hh[t] = f2bf(Whh[t]);
    if (t < n_ih) o_ih[t] = f2bf(Wih[t]);
    if (t < n_w)  { o1[t] = f2bf(W1[t]); o2[t] = f2bf(W2[t]); }
  }
}

// ---------- kernel 3: one workgroup per 16-row strip; LSTM + full 255-step RK4 ----------
// LDS layout (floats), 176KB total:
//   sH   [    0,  8192)  f32 state strip (also holds bf16 LSTM A-staging in phase 0/1)
//   sAbf [ 8192, 12288)  bf16 A-operand strip (dedicated; maintained = bf16(h) between steps)
//   sG   [12288, 45056)  LSTM gates 16x2048 f32; reused in RK4 as:
//        sK1 = +12288, sK2 = +20480, sK3 = +28672, sUbf(bf16) = +36864
__global__ void __launch_bounds__(BLK) ode_lstm_mega(
    const float* __restrict__ inputs, const float* __restrict__ h_in,
    const float* __restrict__ c_in,
    const float* __restrict__ b_ih, const float* __restrict__ b_hh,
    const float* __restrict__ b1,   const float* __restrict__ b2,
    const bf16_t* __restrict__ Wih_bf, const bf16_t* __restrict__ Whh_bf,
    const bf16_t* __restrict__ W1_bf,  const bf16_t* __restrict__ W2_bf,
    const float* __restrict__ dts, const int* __restrict__ snap,
    float* __restrict__ out_h, float* __restrict__ out_c) {
  extern __shared__ float lds[];
  float*  sH   = lds;
  bf16_t* sAbf = (bf16_t*)(lds + 8192);
  float*  sG   = lds + 12288;
  float*  sK1  = lds + 12288;
  float*  sK2  = lds + 20480;
  float*  sK3  = lds + 28672;
  bf16_t* sUbf = (bf16_t*)(lds + 36864);
  bf16_t* sXbf = (bf16_t*)lds;            // 16x128 bf16  (phase 0/1 only)
  bf16_t* sHbf = (bf16_t*)lds + 2048;     // 16x512 bf16  (phase 0/1 only)

  const int strip0 = blockIdx.x * STRIP;
  const int wave   = threadIdx.x >> 5;          // 0..15
  const int lane   = threadIdx.x & 31;
  const int lh     = lane & 15;                 // A row / B column within tile
  const int kb_off = (lane >> 4) << 3;          // 0 or 8
  int snapR[8];
#pragma unroll
  for (int v = 0; v < 8; ++v) snapR[v] = snap[strip0 + v + kb_off];

  // ===== Phase 0: stage LSTM A-operands (inputs, h) as bf16 in LDS, once =====
  {
    const float* xin = inputs + (size_t)strip0 * INDIM;
    const float* hin = h_in  + (size_t)strip0 * HDIM;
    for (int e = threadIdx.x; e < 1024 + 4096; e += BLK) {
      if (e < 1024) {
        v2f x = ((const v2f*)xin)[e];
        ((unsigned*)sXbf)[e] = pack2bf(x[0], x[1]);
      } else {
        int i = e - 1024;
        v2f x = ((const v2f*)hin)[i];
        ((unsigned*)sHbf)[i] = pack2bf(x[0], x[1]);
      }
    }
  }
  __syncthreads();

  // ===== Phase 1: LSTM gates = x@W_ih^T + h@W_hh^T + b  (WMMA) =====
#pragma unroll 1
  for (int it = 0; it < 4; ++it) {              // 128 tile-cols / (16 waves x 2 tiles)
    int c0 = (wave + 16 * (2 * it))     * 16 + lh;
    int c1 = (wave + 16 * (2 * it + 1)) * 16 + lh;
    v8f a0 = zero8(), a1 = zero8();
    gemm_pair(sXbf, INDIM, lh, Wih_bf, INDIM, c0, c1, kb_off, INDIM / 32, a0, a1);
    gemm_pair(sHbf, HDIM,  lh, Whh_bf, HDIM,  c0, c1, kb_off, HDIM / 32,  a0, a1);
    float bi0 = b_ih[c0] + b_hh[c0];
    float bi1 = b_ih[c1] + b_hh[c1];
#pragma unroll
    for (int v = 0; v < 8; ++v) {
      int M = v + kb_off;
      sG[M * 2048 + c0] = a0[v] + bi0;
      sG[M * 2048 + c1] = a1[v] + bi1;
    }
  }
  __syncthreads();

  // ===== Phase 2: LSTM elementwise -> new_c (out), h0 (sH f32 + sAbf bf16), snapshot 0 =====
  for (int e = threadIdx.x; e < STRIP * HDIM; e += BLK) {
    int m = e >> 9, n = e & 511;
    float ig = sG[m * 2048 + n];
    float fg = sG[m * 2048 + n + 512];
    float gg = sG[m * 2048 + n + 1024];
    float og = sG[m * 2048 + n + 1536];
    float cold = c_in[(size_t)(strip0 + m) * HDIM + n];
    float nc = sigm(fg) * cold + sigm(ig) * tanhf(gg);
    float h0 = sigm(og) * tanhf(nc);
    out_c[(size_t)(strip0 + m) * HDIM + n] = nc;
    sH[e]   = h0;
    sAbf[e] = f2bf(h0);
    if (snap[strip0 + m] == 0) out_h[(size_t)(strip0 + m) * HDIM + n] = h0;
  }
  __syncthreads();

  // ===== Phase 3: 255-step RK4, fully inside the workgroup =====
  const float* kprevArr[4] = {nullptr, sK1, sK2, sK3};
  float*       koutArr[3]  = {sK1, sK2, sK3};
#pragma unroll 1
  for (int step = 0; step < BSZ - 1; ++step) {
    float dt = dts[step];
#pragma unroll 1
    for (int stage = 0; stage < 4; ++stage) {
      // --- prepare A = bf16(h + alpha*k_prev), cooperative, once per GEMM ---
      // (stage 0 skipped: sAbf already holds bf16(h) from phase 2 / previous combine)
      if (stage > 0) {
        float al = (stage == 3) ? dt : 0.5f * dt;
        const float* kp = kprevArr[stage];
        for (int e = threadIdx.x; e < 4096; e += BLK) {
          v2f hv = ((const v2f*)sH)[e];
          v2f kv = ((const v2f*)kp)[e];
          ((unsigned*)sAbf)[e] = pack2bf(hv[0] + al * kv[0], hv[1] + al * kv[1]);
        }
        __syncthreads();
      }
      int c0 = wave * 16 + lh;
      int c1 = (wave + 16) * 16 + lh;
      // --- GEMM1: u = tanh(A @ W1^T + b1), written directly as bf16 ---
      {
        v8f a0 = zero8(), a1 = zero8();
        gemm_pair(sAbf, HDIM, lh, W1_bf, HDIM, c0, c1, kb_off, HDIM / 32, a0, a1);
        float bb0 = b1[c0], bb1 = b1[c1];
#pragma unroll
        for (int v = 0; v < 8; ++v) {
          int M = v + kb_off;
          sUbf[M * HDIM + c0] = f2bf(tanhf(a0[v] + bb0));
          sUbf[M * HDIM + c1] = f2bf(tanhf(a1[v] + bb1));
        }
      }
      __syncthreads();
      // --- GEMM2: k = u @ W2^T + b2 ; stage 3 fuses RK4 combine + snapshot ---
      {
        v8f a0 = zero8(), a1 = zero8();
        gemm_pair(sUbf, HDIM, lh, W2_bf, HDIM, c0, c1, kb_off, HDIM / 32, a0, a1);
        float bb0 = b2[c0], bb1 = b2[c1];
        if (stage < 3) {
          float* ko = koutArr[stage];
#pragma unroll
          for (int v = 0; v < 8; ++v) {
            int M = v + kb_off;
            ko[M * HDIM + c0] = a0[v] + bb0;
            ko[M * HDIM + c1] = a1[v] + bb1;
          }
        } else {
          float i6 = dt * (1.0f / 6.0f);
#pragma unroll
          for (int v = 0; v < 8; ++v) {
            int M = v + kb_off;
            int o0 = M * HDIM + c0, o1 = M * HDIM + c1;
            float k40 = a0[v] + bb0, k41 = a1[v] + bb1;
            float hn0 = sH[o0] + i6 * (sK1[o0] + 2.f * sK2[o0] + 2.f * sK3[o0] + k40);
            float hn1 = sH[o1] + i6 * (sK1[o1] + 2.f * sK2[o1] + 2.f * sK3[o1] + k41);
            sH[o0] = hn0;          sH[o1] = hn1;
            sAbf[o0] = f2bf(hn0);  sAbf[o1] = f2bf(hn1);   // keep bf16 mirror current
            if (snapR[v] == step + 1) {
              out_h[(size_t)(strip0 + M) * HDIM + c0] = hn0;
              out_h[(size_t)(strip0 + M) * HDIM + c1] = hn1;
            }
          }
        }
      }
      __syncthreads();
    }
  }
}

// ---------- launch ----------
extern "C" void kernel_launch(void* const* d_in, const int* in_sizes, int n_in,
                              void* d_out, int out_size, void* d_ws, size_t ws_size,
                              hipStream_t stream) {
  const float* inputs = (const float*)d_in[0];
  const float* h_in   = (const float*)d_in[1];
  const float* c_in   = (const float*)d_in[2];
  const float* ts     = (const float*)d_in[3];
  const float* W_ih   = (const float*)d_in[4];
  const float* W_hh   = (const float*)d_in[5];
  const float* b_ih   = (const float*)d_in[6];
  const float* b_hh   = (const float*)d_in[7];
  const float* W1     = (const float*)d_in[8];
  const float* b1     = (const float*)d_in[9];
  const float* W2     = (const float*)d_in[10];
  const float* b2     = (const float*)d_in[11];

  char* ws = (char*)d_ws;
  float*  dts    = (float*)ws;                         // 255 f32
  int*    snap   = (int*)(ws + 1024);                  // 256 i32
  bf16_t* Wih_bf = (bf16_t*)(ws + 2048);               // 2048x128
  bf16_t* Whh_bf = Wih_bf + (size_t)4 * HDIM * INDIM;  // 2048x512
  bf16_t* W1_bf  = Whh_bf + (size_t)4 * HDIM * HDIM;   // 512x512
  bf16_t* W2_bf  = W1_bf + (size_t)HDIM * HDIM;        // 512x512  (~3.5 MB total)

  float* out_h = (float*)d_out;
  float* out_c = out_h + (size_t)BSZ * HDIM;

  sort_ts_kernel<<<1, 256, 0, stream>>>(ts, dts, snap);
  convert_kernel<<<1024, 256, 0, stream>>>(W_ih, W_hh, W1, W2,
                                           Wih_bf, Whh_bf, W1_bf, W2_bf);
  ode_lstm_mega<<<BSZ / STRIP, BLK, 45056 * sizeof(float), stream>>>(
      inputs, h_in, c_in, b_ih, b_hh, b1, b2,
      Wih_bf, Whh_bf, W1_bf, W2_bf, dts, snap, out_h, out_c);

  (void)in_sizes; (void)n_in; (void)out_size; (void)ws_size;
}
